// SessionGraph_24077586661603
// MI455X (gfx1250) — matrix operand order repaired and checked by reading, not compiled, not run on hardware
//
#include <hip/hip_runtime.h>

// GNN gated cell, B=2048, N=H=128, 2 steps.
// One workgroup (8 wave32) per batch element; bf16 WMMA with f32 accum.

#define NN  128
#define HID 128

typedef __attribute__((ext_vector_type(16))) __bf16        v16bf;
typedef __attribute__((ext_vector_type(8)))  float         v8f;
typedef __attribute__((ext_vector_type(8)))  unsigned short v8u;
typedef __attribute__((ext_vector_type(16))) unsigned short v16u;

__device__ __forceinline__ unsigned short f2bf(float f) {
  unsigned int u = __float_as_uint(f);
  u += 0x7FFFu + ((u >> 16) & 1u);   // round-to-nearest-even
  return (unsigned short)(u >> 16);
}
__device__ __forceinline__ float bf2f(unsigned short u) {
  return __uint_as_float(((unsigned int)u) << 16);
}
__device__ __forceinline__ v8f splat8(float x) {
  v8f r;
#pragma unroll
  for (int i = 0; i < 8; ++i) r[i] = x;
  return r;
}
__device__ __forceinline__ v16u cat16(v8u a, v8u b) {
  return __builtin_shufflevector(a, b, 0,1,2,3,4,5,6,7,8,9,10,11,12,13,14,15);
}

// A-fragment (16x32, bf16) from row-major S[M][K] (pitch in elements).
// lanes 0-15: K {0..7} in v0-3, {16..23} in v4-7; lanes 16-31: +8.
__device__ __forceinline__ v16u loadAfrag(const unsigned short* S, int pitch,
                                          int rowbase, int k0, int l15, int hi) {
  const unsigned short* p = S + (rowbase + l15) * pitch + k0 + hi * 8;
  v8u c0 = *(const v8u*)(p);
  v8u c1 = *(const v8u*)(p + 16);
  return cat16(c0, c1);
}
// B-fragment (32x16, bf16) sourced from T = B^T row-major [N][K]:
// lanes 0-15: K {0..7} in v0-3, {16..23} in v4-7; lanes 16-31: +8.
__device__ __forceinline__ v16u loadBfrag(const unsigned short* T, int pitch,
                                          int rowbase, int k0, int l15, int hi) {
  const unsigned short* p = T + (rowbase + l15) * pitch + k0 + hi * 8;
  v8u c0 = *(const v8u*)(p);
  v8u c1 = *(const v8u*)(p + 16);
  return cat16(c0, c1);
}

__device__ __forceinline__ v8f wmma_bf16(v16u a, v16u b, v8f c) {
  return __builtin_amdgcn_wmma_f32_16x16x32_bf16(
      false, __builtin_bit_cast(v16bf, a),
      false, __builtin_bit_cast(v16bf, b),
      (short)0, c, false, false);
}

// ---- weight prep: f32 -> bf16 into workspace ----
// layout (elements): Win 0..16384, Wout ..32768, Whh ..81920, Wih ..180224
__global__ void prep_kernel(const float* __restrict__ W_in,
                            const float* __restrict__ W_out,
                            const float* __restrict__ w_hh,
                            const float* __restrict__ w_ih,
                            unsigned short* __restrict__ ws) {
  int idx = blockIdx.x * 256 + threadIdx.x;
  if (idx < 16384)        ws[idx] = f2bf(W_in[idx]);
  else if (idx < 32768)   ws[idx] = f2bf(W_out[idx - 16384]);
  else if (idx < 81920)   ws[idx] = f2bf(w_hh[idx - 32768]);
  else if (idx < 180224)  ws[idx] = f2bf(w_ih[idx - 81920]);
}

// ---- LDS layout (bytes) ----
#define APITCH 272          // 256 + 16 pad (bf16 elements)
#define HPITCH 144          // 128 + 16 pad
#define OFF_AS     0        // A bf16        128*272*2 = 69632
#define OFF_H0     69632    // h buf0 bf16   128*144*2 = 36864
#define OFF_H1     106496   // h buf1 bf16
#define OFF_HINT   143360   // h_in^T bf16
#define OFF_HOUTT  180224   // h_out^T bf16
#define OFF_INP    217088   // inputs bf16   128*272*2 = 69632
#define OFF_BIAS   286720   // 1280 f32 = 5120
#define SMEM_BYTES 291840
// bias sub-offsets (f32): b_ih 0, b_hh 384, b_iah 768, b_oah 896, b_in 1024, b_out 1152

__global__ __launch_bounds__(256, 1)
void gnn_kernel(const float* __restrict__ Ag, const float* __restrict__ Hg,
                const unsigned short* __restrict__ wbf,
                const float* __restrict__ g_bih, const float* __restrict__ g_bhh,
                const float* __restrict__ g_biah, const float* __restrict__ g_boah,
                const float* __restrict__ g_bin, const float* __restrict__ g_bout,
                float* __restrict__ Out) {
  extern __shared__ char smem[];
  unsigned short* As    = (unsigned short*)(smem + OFF_AS);
  unsigned short* h0    = (unsigned short*)(smem + OFF_H0);
  unsigned short* h1    = (unsigned short*)(smem + OFF_H1);
  unsigned short* hint  = (unsigned short*)(smem + OFF_HINT);
  unsigned short* houtt = (unsigned short*)(smem + OFF_HOUTT);
  unsigned short* inp   = (unsigned short*)(smem + OFF_INP);
  float*          bias  = (float*)(smem + OFF_BIAS);

  const unsigned short* Win  = wbf;
  const unsigned short* Wout = wbf + 16384;
  const unsigned short* Whh  = wbf + 32768;
  const unsigned short* Wih  = wbf + 81920;

  const int tid  = threadIdx.x;
  const int lane = tid & 31;
  const int wv   = tid >> 5;
  const int l15  = lane & 15;
  const int hi   = lane >> 4;
  const int b    = blockIdx.x;

  // ---- stage A (f32 -> bf16) : resident for both steps ----
  {
    int row = tid >> 1, half = tid & 1;
    const float* src = Ag + (size_t)b * (NN * 2 * NN) + row * (2 * NN) + half * NN;
    unsigned short* dst = As + row * APITCH + half * NN;
#pragma unroll
    for (int c = 0; c < 16; ++c) {
      const float4* s4 = (const float4*)(src + c * 8);
      float4 f0 = s4[0], f1 = s4[1];
      v8u o;
      o[0] = f2bf(f0.x); o[1] = f2bf(f0.y); o[2] = f2bf(f0.z); o[3] = f2bf(f0.w);
      o[4] = f2bf(f1.x); o[5] = f2bf(f1.y); o[6] = f2bf(f1.z); o[7] = f2bf(f1.w);
      *(v8u*)(dst + c * 8) = o;
    }
  }
  // ---- stage hidden ----
  {
    int row = tid >> 1, half = tid & 1;
    const float* src = Hg + (size_t)b * (NN * HID) + row * HID + half * 64;
    unsigned short* dst = h0 + row * HPITCH + half * 64;
#pragma unroll
    for (int c = 0; c < 8; ++c) {
      const float4* s4 = (const float4*)(src + c * 8);
      float4 f0 = s4[0], f1 = s4[1];
      v8u o;
      o[0] = f2bf(f0.x); o[1] = f2bf(f0.y); o[2] = f2bf(f0.z); o[3] = f2bf(f0.w);
      o[4] = f2bf(f1.x); o[5] = f2bf(f1.y); o[6] = f2bf(f1.z); o[7] = f2bf(f1.w);
      *(v8u*)(dst + c * 8) = o;
    }
  }
  // ---- biases to LDS (f32) ----
  for (int i = tid; i < 1280; i += 256) {
    float v;
    if (i < 384)       v = g_bih[i];
    else if (i < 768)  v = g_bhh[i - 384];
    else if (i < 896)  v = g_biah[i - 768];
    else if (i < 1024) v = g_boah[i - 896];
    else if (i < 1152) v = g_bin[i - 1024];
    else               v = g_bout[i - 1152];
    bias[i] = v;
  }
  __syncthreads();

  for (int step = 0; step < 2; ++step) {
    const unsigned short* hcur = step ? h1 : h0;

    // ===== Phase 1: h_in = h @ W_in^T + b_in ; h_out = h @ W_out^T + b_out
    // stored transposed: hint[j][m], houtt[j][m]
#pragma unroll 1
    for (int i = 0; i < 16; ++i) {
      int t = wv * 16 + i;
      int which = t >> 6;          // 0 = in, 1 = out
      int mt = (t >> 3) & 7;       // node tile (M)
      int jt = t & 7;              // feature tile (N)
      const unsigned short* W = which ? Wout : Win;
      v8f acc = splat8(bias[(which ? 1152 : 1024) + jt * 16 + l15]);
#pragma unroll
      for (int kt = 0; kt < 4; ++kt) {
        v16u a  = loadAfrag(hcur, HPITCH, mt * 16, kt * 32, l15, hi);
        v16u bb = loadBfrag(W, HID, jt * 16, kt * 32, l15, hi);
        acc = wmma_bf16(a, bb, acc);
      }
      unsigned short* dst =
          (which ? houtt : hint) + (jt * 16 + l15) * HPITCH + mt * 16 + hi * 8;
      v8u o;
#pragma unroll
      for (int r = 0; r < 8; ++r) o[r] = f2bf(acc[r]);
      *(v8u*)dst = o;
    }
    __syncthreads();

    // ===== Phase 2: inputs[n][j]      = (A_in  @ h_in )[n][j] + b_iah[j]
    //                inputs[n][128+j]  = (A_out @ h_out)[n][j] + b_oah[j]
    // computed as C[j][n] = sum_m hT[j][m] * A[n][m], stored as C^T (contiguous)
#pragma unroll 1
    for (int i = 0; i < 16; ++i) {
      int t = wv * 16 + i;
      int which = t >> 6;
      int mt = (t >> 3) & 7;       // feature tile (M = j)
      int nt = t & 7;              // node tile (N = n)
      const unsigned short* HT = which ? houtt : hint;
      int jbase = mt * 16 + hi * 8;
      const float* bv = bias + (which ? 896 : 768);
      v8f acc;
#pragma unroll
      for (int r = 0; r < 8; ++r) acc[r] = bv[jbase + r];
#pragma unroll
      for (int kt = 0; kt < 4; ++kt) {
        v16u a  = loadAfrag(HT, HPITCH, mt * 16, kt * 32, l15, hi);
        v16u bb = loadBfrag(As, APITCH, nt * 16, which * 128 + kt * 32, l15, hi);
        acc = wmma_bf16(a, bb, acc);
      }
      unsigned short* dst =
          inp + (nt * 16 + l15) * APITCH + which * 128 + mt * 16 + hi * 8;
      v8u o;
#pragma unroll
      for (int r = 0; r < 8; ++r) o[r] = f2bf(acc[r]);
      *(v8u*)dst = o;
    }
    __syncthreads();

    // ===== Phase 3: gi = inputs @ w_ih^T + b_ih ; gh = h @ w_hh^T + b_hh ; gates
    // wave wv handles node tile nt=wv, all 8 feature tiles; 6 acc tiles in regs
#pragma unroll 1
    for (int jt = 0; jt < 8; ++jt) {
      const int nt = wv;
      const int g0 = jt * 16 + l15;
      v8f a_ir = splat8(bias[g0]);
      v8f a_ii = splat8(bias[128 + g0]);
      v8f a_in = splat8(bias[256 + g0]);
      v8f a_hr = splat8(bias[384 + g0]);
      v8f a_hi = splat8(bias[512 + g0]);
      v8f a_hn = splat8(bias[640 + g0]);
#pragma unroll
      for (int kt = 0; kt < 8; ++kt) {           // K = 256 over inputs
        v16u a  = loadAfrag(inp, APITCH, nt * 16, kt * 32, l15, hi);
        v16u b0 = loadBfrag(Wih, 256, jt * 16, kt * 32, l15, hi);
        a_ir = wmma_bf16(a, b0, a_ir);
        v16u b1 = loadBfrag(Wih, 256, 128 + jt * 16, kt * 32, l15, hi);
        a_ii = wmma_bf16(a, b1, a_ii);
        v16u b2 = loadBfrag(Wih, 256, 256 + jt * 16, kt * 32, l15, hi);
        a_in = wmma_bf16(a, b2, a_in);
      }
#pragma unroll
      for (int kt = 0; kt < 4; ++kt) {           // K = 128 over h
        v16u a  = loadAfrag(hcur, HPITCH, nt * 16, kt * 32, l15, hi);
        v16u b0 = loadBfrag(Whh, HID, jt * 16, kt * 32, l15, hi);
        a_hr = wmma_bf16(a, b0, a_hr);
        v16u b1 = loadBfrag(Whh, HID, 128 + jt * 16, kt * 32, l15, hi);
        a_hi = wmma_bf16(a, b1, a_hi);
        v16u b2 = loadBfrag(Whh, HID, 256 + jt * 16, kt * 32, l15, hi);
        a_hn = wmma_bf16(a, b2, a_hn);
      }
      // GRU-style gates, entirely in registers
#pragma unroll
      for (int r = 0; r < 8; ++r) {
        int n = nt * 16 + hi * 8 + r;
        int j = jt * 16 + l15;
        float rg = 1.f / (1.f + __expf(-(a_ir[r] + a_hr[r])));
        float ig = 1.f / (1.f + __expf(-(a_ii[r] + a_hi[r])));
        float ng = tanhf(a_in[r] + rg * a_hn[r]);
        float hv = bf2f(hcur[n * HPITCH + j]);
        float ov = ng + ig * (hv - ng);
        if (step == 0) h1[n * HPITCH + j] = f2bf(ov);
        else Out[(size_t)b * (NN * HID) + n * HID + j] = ov;
      }
    }
    __syncthreads();
  }
}

extern "C" void kernel_launch(void* const* d_in, const int* in_sizes, int n_in,
                              void* d_out, int out_size, void* d_ws, size_t ws_size,
                              hipStream_t stream) {
  const float* A     = (const float*)d_in[0];
  const float* Hg    = (const float*)d_in[1];
  const float* w_ih  = (const float*)d_in[2];
  const float* w_hh  = (const float*)d_in[3];
  const float* b_ih  = (const float*)d_in[4];
  const float* b_hh  = (const float*)d_in[5];
  const float* b_iah = (const float*)d_in[6];
  const float* b_oah = (const float*)d_in[7];
  const float* W_in  = (const float*)d_in[8];
  const float* b_in  = (const float*)d_in[9];
  const float* W_out = (const float*)d_in[10];
  const float* b_out = (const float*)d_in[11];
  unsigned short* ws = (unsigned short*)d_ws;
  float* out = (float*)d_out;

  // convert all weights to bf16 once per call (deterministic)
  prep_kernel<<<704, 256, 0, stream>>>(W_in, W_out, w_hh, w_ih, ws);

  hipFuncSetAttribute((const void*)gnn_kernel,
                      hipFuncAttributeMaxDynamicSharedMemorySize, SMEM_BYTES);
  gnn_kernel<<<2048, 256, SMEM_BYTES, stream>>>(A, Hg, ws, b_ih, b_hh, b_iah,
                                                b_oah, b_in, b_out, out);
}
